// PropModule_49615462203556
// MI455X (gfx1250) — matrix operand order repaired and checked by reading, not compiled, not run on hardware
//
#include <hip/hip_runtime.h>

// ---------------------------------------------------------------------------
// Interaction-network propagation for MI455X (gfx1250, wave32, WMMA).
// Dense one-hot Rr/Rs are decoded to index arrays; all GEMMs run through
// v_wmma_f32_16x16x32_f16 (f16 in, f32 accumulate).
// ---------------------------------------------------------------------------

typedef __attribute__((ext_vector_type(16))) _Float16 v16h;
typedef __attribute__((ext_vector_type(8)))  _Float16 v8h;
typedef __attribute__((ext_vector_type(8)))  float    v8f;

#define Bb      8
#define Nn      512
#define NRELn   4096
#define IN_DIMn 64
#define REL_DIMn 32
#define NF_PARTn 128
#define NF_RELn  256
#define NF_EFFn  128
#define OUT_DIMn 64
#define PSTEPn   3

#define NEDGE (Bb * NRELn)     // 32768
#define NPART (Bb * Nn)        // 4096

// ---------------------------------------------------------------------------
// Decode one-hot incidence matrix [B, N, NREL] -> index array [B*NREL].
// Threads over edges; loop over N is coalesced across lanes (stride NREL).
// ---------------------------------------------------------------------------
__global__ void k_decode(const float* __restrict__ R, int* __restrict__ idx) {
    int t = blockIdx.x * blockDim.x + threadIdx.x;
    if (t >= NEDGE) return;
    int b = t / NRELn, e = t - b * NRELn;
    const float* p = R + (size_t)b * Nn * NRELn + e;
    int r = 0;
    for (int n = 0; n < Nn; ++n)
        if (p[(size_t)n * NRELn] > 0.5f) r = n;
    idx[t] = r;
}

__global__ void k_cvt_f16(const float* __restrict__ src,
                          _Float16* __restrict__ dst, int n) {
    int t = blockIdx.x * blockDim.x + threadIdx.x;
    if (t < n) dst[t] = (_Float16)src[t];
}

// W [K,N] fp32 row-major -> Wt [N,K] f16 row-major (transposed for GEMM B loads)
__global__ void k_cvt_wt(const float* __restrict__ W, _Float16* __restrict__ Wt,
                         int K, int N) {
    int t = blockIdx.x * blockDim.x + threadIdx.x;
    if (t >= K * N) return;
    int k = t / N, n = t - k * N;
    Wt[(size_t)n * K + k] = (_Float16)W[t];
}

// xrel [NEDGE, 160] = [state16[recv] | state16[send] | f16(Ra)]
__global__ void k_build_xrel(const _Float16* __restrict__ state16,
                             const float* __restrict__ Ra,
                             const int* __restrict__ recv,
                             const int* __restrict__ send,
                             _Float16* __restrict__ xrel) {
    int t = blockIdx.x * blockDim.x + threadIdx.x;
    if (t >= NEDGE * 160) return;
    int row = t / 160, c = t - row * 160;
    int b = row / NRELn;
    _Float16 v;
    if (c < 64) {
        v = state16[((size_t)b * Nn + recv[row]) * IN_DIMn + c];
    } else if (c < 128) {
        v = state16[((size_t)b * Nn + send[row]) * IN_DIMn + (c - 64)];
    } else {
        v = (_Float16)Ra[(size_t)row * REL_DIMn + (c - 128)];
    }
    xrel[(size_t)row * 160 + c] = v;
}

// Gather particle effects into xcat_rp[:, 256:384] (recv) and [:,384:512] (send)
__global__ void k_gather_eff(const _Float16* __restrict__ peff,
                             const int* __restrict__ recv,
                             const int* __restrict__ send,
                             _Float16* __restrict__ xcat_rp) {
    int t = blockIdx.x * blockDim.x + threadIdx.x;
    if (t >= NEDGE * NF_EFFn) return;
    int row = t / NF_EFFn, c = t - row * NF_EFFn;
    int b = row / NRELn;
    xcat_rp[(size_t)row * 512 + 256 + c] =
        peff[((size_t)b * Nn + recv[row]) * NF_EFFn + c];
    xcat_rp[(size_t)row * 512 + 384 + c] =
        peff[((size_t)b * Nn + send[row]) * NF_EFFn + c];
}

// Scatter-add relation effects to particles (fp32 atomics)
__global__ void k_scatter(const _Float16* __restrict__ rel,
                          const int* __restrict__ recv,
                          float* __restrict__ agg) {
    int t = blockIdx.x * blockDim.x + threadIdx.x;
    if (t >= NEDGE * NF_EFFn) return;
    int row = t / NF_EFFn, c = t - row * NF_EFFn;
    int b = row / NRELn;
    atomicAdd(&agg[((size_t)b * Nn + recv[row]) * NF_EFFn + c], (float)rel[t]);
}

// agg fp32 [NPART,128] -> f16 into xcat_pp[:, 128:256]
__global__ void k_cvt_agg(const float* __restrict__ agg,
                          _Float16* __restrict__ xcat_pp) {
    int t = blockIdx.x * blockDim.x + threadIdx.x;
    if (t >= NPART * NF_EFFn) return;
    int m = t / NF_EFFn, c = t - m * NF_EFFn;
    xcat_pp[(size_t)m * 256 + 128 + c] = (_Float16)agg[t];
}

// ---------------------------------------------------------------------------
// WMMA GEMM: out[m, ocol+n] = act(A[m,0:K] @ Wt[n,0:K]^T + bias[n])
//   A  : f16 [M, lda]  (row-major, uses first K columns)
//   Wt : f16 [N, K]    (transposed weights, row-major)
// Each wave computes a 32(M) x 64(N) tile: 2 A-frags x 4 B-frags reused
// across 8 v_wmma_f32_16x16x32_f16 per 32-wide K step.
// ---------------------------------------------------------------------------
__global__ __launch_bounds__(256)
void gemm_wmma(const _Float16* __restrict__ A, int lda,
               const _Float16* __restrict__ Wt,
               const float* __restrict__ bias,
               _Float16* __restrict__ out16, float* __restrict__ out32,
               int ldo, int ocol, int M, int N, int K, int relu) {
    const int lane = threadIdx.x & 31;
    const int wave = threadIdx.x >> 5;
    const int tilesN = N >> 6;
    int tile = blockIdx.x * 8 + wave;
    int tm = tile / tilesN;
    int tn = tile - tm * tilesN;
    if (tm * 32 >= M) return;

    const int l16 = lane & 15;
    const int hi  = lane >> 4;   // 0 for lanes 0-15, 1 for lanes 16-31

    v8f c[2][4];
#pragma unroll
    for (int i = 0; i < 2; ++i)
#pragma unroll
        for (int j = 0; j < 4; ++j) c[i][j] = (v8f){};

    const _Float16* arow0 = A + (size_t)(tm * 32 + l16) * lda;
    const _Float16* arow1 = arow0 + (size_t)16 * lda;
    const _Float16* brow[4];
#pragma unroll
    for (int j = 0; j < 4; ++j)
        brow[j] = Wt + (size_t)(tn * 64 + j * 16 + l16) * K + hi * 16;

    for (int kk = 0; kk < K; kk += 32) {
        // A fragment (16x32 f16): lane holds row m=l16; elems i -> K=kk+16*hi+i
        // (contiguous, matching B)
        v8h a0lo = *(const v8h*)(arow0 + kk + hi * 16);
        v8h a0hi = *(const v8h*)(arow0 + kk + hi * 16 + 8);
        v8h a1lo = *(const v8h*)(arow1 + kk + hi * 16);
        v8h a1hi = *(const v8h*)(arow1 + kk + hi * 16 + 8);
        v16h a0 = __builtin_shufflevector(a0lo, a0hi, 0,1,2,3,4,5,6,7,
                                          8,9,10,11,12,13,14,15);
        v16h a1 = __builtin_shufflevector(a1lo, a1hi, 0,1,2,3,4,5,6,7,
                                          8,9,10,11,12,13,14,15);
        // B fragments (32x16 f16): lane holds col n=l16; elems i -> K=kk+16*hi+i
        v16h b0 = *(const v16h*)(brow[0] + kk);
        v16h b1 = *(const v16h*)(brow[1] + kk);
        v16h b2 = *(const v16h*)(brow[2] + kk);
        v16h b3 = *(const v16h*)(brow[3] + kk);

        c[0][0] = __builtin_amdgcn_wmma_f32_16x16x32_f16(false, a0, false, b0,
                                                         (short)0, c[0][0], false, false);
        c[0][1] = __builtin_amdgcn_wmma_f32_16x16x32_f16(false, a0, false, b1,
                                                         (short)0, c[0][1], false, false);
        c[0][2] = __builtin_amdgcn_wmma_f32_16x16x32_f16(false, a0, false, b2,
                                                         (short)0, c[0][2], false, false);
        c[0][3] = __builtin_amdgcn_wmma_f32_16x16x32_f16(false, a0, false, b3,
                                                         (short)0, c[0][3], false, false);
        c[1][0] = __builtin_amdgcn_wmma_f32_16x16x32_f16(false, a1, false, b0,
                                                         (short)0, c[1][0], false, false);
        c[1][1] = __builtin_amdgcn_wmma_f32_16x16x32_f16(false, a1, false, b1,
                                                         (short)0, c[1][1], false, false);
        c[1][2] = __builtin_amdgcn_wmma_f32_16x16x32_f16(false, a1, false, b2,
                                                         (short)0, c[1][2], false, false);
        c[1][3] = __builtin_amdgcn_wmma_f32_16x16x32_f16(false, a1, false, b3,
                                                         (short)0, c[1][3], false, false);
    }

    // Epilogue: C layout — VGPR v <-> M = v + 8*hi within each 16x16 tile
#pragma unroll
    for (int i = 0; i < 2; ++i) {
        int mbase = tm * 32 + i * 16 + hi * 8;
#pragma unroll
        for (int j = 0; j < 4; ++j) {
            int n = tn * 64 + j * 16 + l16;
            float bv = bias ? bias[n] : 0.0f;
#pragma unroll
            for (int v = 0; v < 8; ++v) {
                float x = c[i][j][v] + bv;
                if (relu) x = fmaxf(x, 0.0f);
                size_t o = (size_t)(mbase + v) * ldo + ocol + n;
                if (out16) out16[o] = (_Float16)x;
                if (out32) out32[o] = x;
            }
        }
    }
}

// ---------------------------------------------------------------------------
// Host side
// ---------------------------------------------------------------------------
static void launch_gemm(const _Float16* A, int lda, const _Float16* Wt,
                        const float* bias, _Float16* o16, float* o32,
                        int ldo, int ocol, int M, int N, int K, int relu,
                        hipStream_t s) {
    int tiles = (M / 32) * (N / 64);
    int blocks = (tiles + 7) / 8;
    gemm_wmma<<<blocks, 256, 0, s>>>(A, lda, Wt, bias, o16, o32, ldo, ocol,
                                     M, N, K, relu);
}

static inline int gdiv(int n, int b) { return (n + b - 1) / b; }

extern "C" void kernel_launch(void* const* d_in, const int* in_sizes, int n_in,
                              void* d_out, int out_size, void* d_ws, size_t ws_size,
                              hipStream_t stream) {
    const float* state = (const float*)d_in[0];
    const float* Rr    = (const float*)d_in[1];
    const float* Rs    = (const float*)d_in[2];
    const float* Ra    = (const float*)d_in[3];
    const float* W[10] = {(const float*)d_in[4],  (const float*)d_in[6],
                          (const float*)d_in[8],  (const float*)d_in[10],
                          (const float*)d_in[12], (const float*)d_in[14],
                          (const float*)d_in[16], (const float*)d_in[18],
                          (const float*)d_in[20], (const float*)d_in[22]};
    const float* Bv[10] = {(const float*)d_in[5],  (const float*)d_in[7],
                           (const float*)d_in[9],  (const float*)d_in[11],
                           (const float*)d_in[13], (const float*)d_in[15],
                           (const float*)d_in[17], (const float*)d_in[19],
                           (const float*)d_in[21], (const float*)d_in[23]};
    // weight dims (K = fan_in, N = fan_out)
    const int WK[10] = {64, 128, 128, 160, 256, 256, 512, 256, 128, 128};
    const int WN[10] = {128, 128, 128, 256, 256, 256, 128, 128, 128, 64};

    // workspace carve-up (256B aligned)
    char* ws = (char*)d_ws;
    size_t off = 0;
    auto carve = [&](size_t bytes) -> char* {
        char* p = ws + off;
        off = (off + bytes + 255) & ~(size_t)255;
        return p;
    };
    int*      recv    = (int*)carve(NEDGE * 4);
    int*      send    = (int*)carve(NEDGE * 4);
    _Float16* state16 = (_Float16*)carve((size_t)NPART * IN_DIMn * 2);
    _Float16* Wt[10];
    for (int i = 0; i < 10; ++i)
        Wt[i] = (_Float16*)carve((size_t)WK[i] * WN[i] * 2);
    _Float16* xrel    = (_Float16*)carve((size_t)NEDGE * 160 * 2);
    _Float16* hrel_a  = (_Float16*)carve((size_t)NEDGE * 256 * 2);
    _Float16* hrel_b  = (_Float16*)carve((size_t)NEDGE * 256 * 2);
    _Float16* xcat_rp = (_Float16*)carve((size_t)NEDGE * 512 * 2);
    _Float16* releff  = (_Float16*)carve((size_t)NEDGE * 128 * 2);
    _Float16* xcat_pp = (_Float16*)carve((size_t)NPART * 256 * 2);
    _Float16* hpart_a = (_Float16*)carve((size_t)NPART * 128 * 2);
    _Float16* hpart_b = (_Float16*)carve((size_t)NPART * 128 * 2);
    float*    agg32   = (float*)carve((size_t)NPART * 128 * 4);
    _Float16* peff    = (_Float16*)carve((size_t)NPART * 128 * 2);
    _Float16* hfin    = (_Float16*)carve((size_t)NPART * 128 * 2);
    (void)ws_size; (void)n_in; (void)in_sizes; (void)out_size;

    // --- decode incidence matrices to indices ---
    k_decode<<<gdiv(NEDGE, 256), 256, 0, stream>>>(Rr, recv);
    k_decode<<<gdiv(NEDGE, 256), 256, 0, stream>>>(Rs, send);

    // --- precision conversion: state + weights ---
    k_cvt_f16<<<gdiv(NPART * IN_DIMn, 256), 256, 0, stream>>>(state, state16,
                                                              NPART * IN_DIMn);
    for (int i = 0; i < 10; ++i)
        k_cvt_wt<<<gdiv(WK[i] * WN[i], 256), 256, 0, stream>>>(W[i], Wt[i],
                                                               WK[i], WN[i]);

    // --- relation input concat ---
    k_build_xrel<<<gdiv(NEDGE * 160, 256), 256, 0, stream>>>(state16, Ra, recv,
                                                             send, xrel);

    // --- particle encoder (writes into xcat_pp[:, 0:128]) ---
    launch_gemm(state16, 64,  Wt[0], Bv[0], hpart_a, nullptr, 128, 0,
                NPART, 128, 64, 1, stream);
    launch_gemm(hpart_a, 128, Wt[1], Bv[1], hpart_b, nullptr, 128, 0,
                NPART, 128, 128, 1, stream);
    launch_gemm(hpart_b, 128, Wt[2], Bv[2], xcat_pp, nullptr, 256, 0,
                NPART, 128, 128, 1, stream);

    // --- relation encoder (writes into xcat_rp[:, 0:256]) ---
    launch_gemm(xrel,   160, Wt[3], Bv[3], hrel_a, nullptr, 256, 0,
                NEDGE, 256, 160, 1, stream);
    launch_gemm(hrel_a, 256, Wt[4], Bv[4], hrel_b, nullptr, 256, 0,
                NEDGE, 256, 256, 1, stream);
    launch_gemm(hrel_b, 256, Wt[5], Bv[5], xcat_rp, nullptr, 512, 0,
                NEDGE, 256, 256, 1, stream);

    // --- propagation ---
    hipMemsetAsync(peff, 0, (size_t)NPART * 128 * 2, stream);
    for (int step = 0; step < PSTEPn; ++step) {
        k_gather_eff<<<gdiv(NEDGE * 128, 256), 256, 0, stream>>>(peff, recv,
                                                                 send, xcat_rp);
        hipMemsetAsync(agg32, 0, (size_t)NPART * 128 * 4, stream);
        launch_gemm(xcat_rp, 512, Wt[6], Bv[6], releff, nullptr, 128, 0,
                    NEDGE, 128, 512, 1, stream);
        k_scatter<<<gdiv(NEDGE * 128, 256), 256, 0, stream>>>(releff, recv,
                                                              agg32);
        k_cvt_agg<<<gdiv(NPART * 128, 256), 256, 0, stream>>>(agg32, xcat_pp);
        launch_gemm(xcat_pp, 256, Wt[7], Bv[7], peff, nullptr, 128, 0,
                    NPART, 128, 256, 1, stream);
    }

    // --- output head: pr1 writes fp32 directly to d_out, no relu ---
    launch_gemm(peff, 128, Wt[8], Bv[8], hfin, nullptr, 128, 0,
                NPART, 128, 128, 1, stream);
    launch_gemm(hfin, 128, Wt[9], Bv[9], nullptr, (float*)d_out, 64, 0,
                NPART, 64, 128, 0, stream);
}
